// MPNN_54443005444256
// MI455X (gfx1250) — matrix operand order repaired and checked by reading, not compiled, not run on hardware
//
#include <hip/hip_runtime.h>

typedef _Float16 half_t;
typedef __attribute__((ext_vector_type(16))) _Float16 v16h;
typedef __attribute__((ext_vector_type(8)))  _Float16 v8h;
typedef __attribute__((ext_vector_type(8)))  float    v8f;
typedef unsigned int v4u __attribute__((ext_vector_type(4)));
typedef int v8i_t __attribute__((ext_vector_type(8)));
typedef int v4i_t __attribute__((ext_vector_type(4)));

// ---------------- problem constants ----------------
constexpr int kN = 131072, kE = 262144, kB = 4096;
constexpr int kND = 16, kH = 64;

// ---------------- workspace layout (bytes, all 4KB aligned) ----------------
constexpr size_t OFF_W2R  = 0;                                   // 133120 halfs (w2 frags + b_e2 frags)
constexpr size_t OFF_ROOT = OFF_W2R  + 266240;                   // 2048 halfs
constexpr size_t OFF_GWIH = OFF_ROOT + 4096;                     // 12288 halfs
constexpr size_t OFF_GWHH = OFF_GWIH + 24576;                    // 12288 halfs (contiguous after GWIH for TDM)
constexpr size_t OFF_LWIH = OFF_GWHH + 24576;                    // 32768 halfs
constexpr size_t OFF_LWHH = OFF_LWIH + 65536;                    // 16384 halfs
constexpr size_t OFF_FC1  = OFF_LWHH + 32768;                    // 8192 halfs
constexpr size_t OFF_MSUM = OFF_FC1  + 16384;                    // N*64 f32
constexpr size_t OFF_CNT  = OFF_MSUM + (size_t)kN*64*4;          // N f32
constexpr size_t OFF_H32  = OFF_CNT  + (size_t)kN*4;             // N*64 f32
constexpr size_t OFF_H16  = OFF_H32  + (size_t)kN*64*4;          // N*64 f16
constexpr size_t OFF_HL32 = OFF_H16  + (size_t)kN*64*2;          // B*64 f32
constexpr size_t OFF_CL   = OFF_HL32 + (size_t)kB*64*4;          // B*64 f32
constexpr size_t OFF_HL16 = OFF_CL   + (size_t)kB*64*4;          // B*64 f16
constexpr size_t OFF_QS16 = OFF_HL16 + (size_t)kB*64*2;          // B*128 f16
constexpr size_t OFF_EB   = OFF_QS16 + (size_t)kB*128*2;         // N f32
constexpr size_t OFF_EX   = OFF_EB   + (size_t)kN*4;             // N f32
constexpr size_t OFF_M    = OFF_EX   + (size_t)kN*4;             // B f32
constexpr size_t OFF_DEN  = OFF_M    + (size_t)kB*4;             // B f32
constexpr size_t OFF_RG   = OFF_DEN  + (size_t)kB*4;             // B*64 f32

// ---------------- helpers ----------------
__device__ __forceinline__ v16h cat16(v8h lo, v8h hi) {
  v16h r;
#pragma unroll
  for (int i = 0; i < 8; ++i) { r[i] = lo[i]; r[i + 8] = hi[i]; }
  return r;
}
__device__ __forceinline__ v8f splat8(float v) {
  v8f r;
#pragma unroll
  for (int i = 0; i < 8; ++i) r[i] = v;
  return r;
}
__device__ __forceinline__ v8f wmma_f16(v16h a, v16h b, v8f c) {
  return __builtin_amdgcn_wmma_f32_16x16x32_f16(false, a, false, b, (short)0, c, false, false);
}
__device__ __forceinline__ float sigmoidf(float x) { return 1.f / (1.f + __expf(-x)); }
__device__ __forceinline__ void atomicMaxF(float* addr, float val) {
  if (val >= 0.f) atomicMax((int*)addr, __float_as_int(val));
  else            atomicMin((unsigned int*)addr, (unsigned int)__float_as_int(val));
}

// ================= weight packing =================
// B-fragment layout (16x16x32 f16 WMMA, wave32): frag f=(ks*NT+nt);
// lane: n = nt*16 + (lane&15); half j -> k = ks*32 + 8*(lane>>4) + (j&7) + 16*(j>>3).
__global__ void pack_b_frags(const float* __restrict__ src, half_t* __restrict__ dst,
                             int ksteps, int ntiles, int Kreal, int Nreal, int ld, int trans) {
  int idx = blockIdx.x * blockDim.x + threadIdx.x;
  int total = ksteps * ntiles * 512;
  if (idx >= total) return;
  int j = idx & 15, lane = (idx >> 4) & 31;
  int f = idx >> 9;
  int nt = f % ntiles, ks = f / ntiles;
  int k = ks * 32 + 8 * (lane >> 4) + (j & 7) + 16 * (j >> 3);
  int n = nt * 16 + (lane & 15);
  float v = 0.f;
  if (k < Kreal && n < Nreal) v = trans ? src[(size_t)n * ld + k] : src[(size_t)k * ld + n];
  dst[idx] = (half_t)v;
}

// w_e2 [128,1024] -> per-d (d=0..15) B blocks [K=128][N=64], then b_e2 block (K pad 32).
__global__ void pack_w2r_kernel(const float* __restrict__ w_e2, const float* __restrict__ b_e2,
                                half_t* __restrict__ dst) {
  int idx = blockIdx.x * blockDim.x + threadIdx.x;
  if (idx >= 133120) return;
  int j = idx & 15, lane = (idx >> 4) & 31;
  float v;
  if (idx < 131072) {
    int frag = idx >> 9;                    // 0..255
    int d = frag >> 4, ks = (frag >> 2) & 3, nt = frag & 3;
    int k = ks * 32 + 8 * (lane >> 4) + (j & 7) + 16 * (j >> 3);
    int n = nt * 16 + (lane & 15);
    v = w_e2[(size_t)k * 1024 + d * 64 + n];
  } else {
    int nt = (idx - 131072) >> 9;           // 0..3
    int k = 8 * (lane >> 4) + (j & 7) + 16 * (j >> 3);
    int n = nt * 16 + (lane & 15);
    v = (k < 16) ? b_e2[k * 64 + n] : 0.f;
  }
  dst[idx] = (half_t)v;
}

__global__ void fill_f32_kernel(float* __restrict__ p, float v, int n) {
  int i = blockIdx.x * blockDim.x + threadIdx.x;
  if (i < n) p[i] = v;
}

// ================= NNConv: fused edge MLP + per-edge matvec + scatter =================
// wave per 16-edge tile, 4 tiles per wave; msg accumulated across d in WMMA C regs.
// d-loop deliberately NOT unrolled: keeps ~16 B-fragments in flight (no spills),
// B streams from L2 (w2r table is 266KB, L2-resident).
__global__ __launch_bounds__(128) void edge_msg_kernel(
    const float* __restrict__ x, const float* __restrict__ edge_attr,
    const int* __restrict__ edge_index, const float* __restrict__ w_e1,
    const float* __restrict__ b_e1, const half_t* __restrict__ w2r,
    float* __restrict__ msum, float* __restrict__ cnt) {
  __shared__ __align__(16) half_t hid_lds[4][16][128];
  __shared__ __align__(16) half_t x_lds[4][16][16];
  __shared__ int dst_lds[4][16];
  const int wave = threadIdx.x >> 5, lane = threadIdx.x & 31;
  const int hi2 = lane >> 4, m = lane & 15;
  const int wslot = blockIdx.x * 4 + wave;           // 0..4095
  const v16h* bf = (const v16h*)w2r;

  for (int it = 0; it < 4; ++it) {
    const int tile = wslot * 4 + it;                 // 0..16383
    const int ebase = tile * 16;
    __syncthreads();
    if (lane < 16) dst_lds[wave][lane] = edge_index[kE + ebase + lane];
    {
      const int e = ebase + m;
      const int srcn = edge_index[e];
      // x row half -> LDS (f16)
      const float* xr = x + (size_t)srcn * kND + hi2 * 8;
      v8h xa;
#pragma unroll
      for (int j = 0; j < 8; ++j) xa[j] = (half_t)xr[j];
      *(v8h*)&x_lds[wave][m][hi2 * 8] = xa;
      // hid = relu(ea @ w_e1 + b_e1), this lane covers 64 of 128 k's
      const float* eap = edge_attr + (size_t)e * 4;
      const float ea0 = eap[0], ea1 = eap[1], ea2 = eap[2], ea3 = eap[3];
      const int k0 = hi2 * 64;
#pragma unroll 8
      for (int k = 0; k < 64; ++k) {
        const int kk = k0 + k;
        float a = b_e1[kk] + ea0 * w_e1[kk] + ea1 * w_e1[128 + kk] +
                  ea2 * w_e1[256 + kk] + ea3 * w_e1[384 + kk];
        hid_lds[wave][m][kk] = (half_t)fmaxf(a, 0.f);
      }
    }
    __syncthreads();
    // A-fragments: hid (4 k-steps)
    v16h hidf[4];
#pragma unroll
    for (int ks = 0; ks < 4; ++ks) {
      v8h lo = *(const v8h*)&hid_lds[wave][m][ks * 32 + 8 * hi2];
      v8h hh = *(const v8h*)&hid_lds[wave][m][ks * 32 + 16 + 8 * hi2];
      hidf[ks] = cat16(lo, hh);
    }
    v8f c0 = {}, c1 = {}, c2 = {}, c3 = {};
#pragma unroll 1
    for (int d = 0; d < 16; ++d) {
      const half_t s = x_lds[wave][m][d];            // scalar from LDS (no dyn vec extract)
#pragma unroll
      for (int ks = 0; ks < 4; ++ks) {
        v16h af = hidf[ks] * s;                      // v_pk_mul_f16 x8
        const int base = ((d * 4 + ks) * 4) * 32 + lane;
        c0 = wmma_f16(af, bf[base +  0], c0);
        c1 = wmma_f16(af, bf[base + 32], c1);
        c2 = wmma_f16(af, bf[base + 64], c2);
        c3 = wmma_f16(af, bf[base + 96], c3);
      }
    }
    { // b_e2 term: A = [x_row | 0] (K padded to 32)
      v8h xh = *(const v8h*)&x_lds[wave][m][8 * hi2];
      v16h af = {};
#pragma unroll
      for (int j = 0; j < 8; ++j) af[j] = xh[j];
      const int base = 256 * 32 + lane;
      c0 = wmma_f16(af, bf[base +  0], c0);
      c1 = wmma_f16(af, bf[base + 32], c1);
      c2 = wmma_f16(af, bf[base + 64], c2);
      c3 = wmma_f16(af, bf[base + 96], c3);
    }
    // scatter-add msg into msum[dst]
#pragma unroll
    for (int i = 0; i < 8; ++i) {
      const int M = i + 8 * hi2;
      const int dn = dst_lds[wave][M];
      float* row = msum + (size_t)dn * kH + m;
      atomicAdd(row +  0, c0[i]);
      atomicAdd(row + 16, c1[i]);
      atomicAdd(row + 32, c2[i]);
      atomicAdd(row + 48, c3[i]);
    }
    if (lane < 16) atomicAdd(&cnt[dst_lds[wave][lane]], 1.0f);
  }
}

// ================= node update: out = relu(msum/cnt + x@root + bias) =================
__global__ __launch_bounds__(256) void node_update_kernel(
    const float* __restrict__ x, const float* __restrict__ msum, const float* __restrict__ cnt,
    const half_t* __restrict__ rootf, const float* __restrict__ conv_bias,
    float* __restrict__ h32, half_t* __restrict__ h16) {
  const int wave = threadIdx.x >> 5, lane = threadIdx.x & 31;
  const int hi2 = lane >> 4, m = lane & 15;
  const int tile = blockIdx.x * 8 + wave;            // 8192 tiles
  v16h af = {};
  const float* xr = x + (size_t)(tile * 16 + m) * kND + 8 * hi2;
#pragma unroll
  for (int j = 0; j < 8; ++j) af[j] = (half_t)xr[j];
  const v16h* bf = (const v16h*)rootf;
  v8f c[4];
#pragma unroll
  for (int nt = 0; nt < 4; ++nt) {
    c[nt] = splat8(conv_bias[nt * 16 + m]);
    c[nt] = wmma_f16(af, bf[nt * 32 + lane], c[nt]);
  }
#pragma unroll
  for (int i = 0; i < 8; ++i) {
    const int node = tile * 16 + i + 8 * hi2;
    const float cv = fmaxf(cnt[node], 1.0f);
#pragma unroll
    for (int nt = 0; nt < 4; ++nt) {
      const size_t idx = (size_t)node * kH + nt * 16 + m;
      const float v = fmaxf(c[nt][i] + msum[idx] / cv, 0.f);
      h32[idx] = v; h16[idx] = (half_t)v;
    }
  }
}

// ================= one GRU step (in-place on h), weights staged via TDM =================
__global__ __launch_bounds__(256) void gru_step_kernel(
    float* __restrict__ h32, half_t* __restrict__ h16,
    const half_t* __restrict__ wihf, const float* __restrict__ bih,
    const float* __restrict__ bhh) {
  __shared__ __align__(16) half_t wlds[24576];       // 48KB: wih frags then whh frags
#if defined(__gfx1250__) && __has_builtin(__builtin_amdgcn_tensor_load_to_lds) && __has_builtin(__builtin_amdgcn_s_wait_tensorcnt)
  if (threadIdx.x == 0) {
    unsigned long long ga = (unsigned long long)(uintptr_t)wihf;
    unsigned ldsoff = (unsigned)(uintptr_t)(&wlds[0]);
    v4u g0;
    g0[0] = 1u;                                       // count=1
    g0[1] = ldsoff;                                   // lds_addr
    g0[2] = (unsigned)(ga & 0xffffffffu);             // global_addr lo
    g0[3] = (unsigned)((ga >> 32) & 0x1ffffffu) | 0x80000000u;  // addr hi | type=2
    v8i_t g1;
    g1[0] = (int)(3u << 16);                          // data_size = 8B
    g1[1] = (int)(6144u << 16);                       // tensor_dim0[15:0] (49152B/8)
    g1[2] = (int)((6144u >> 16) | (1u << 16));        // dim0 hi | tensor_dim1=1
    g1[3] = (int)(6144u << 16);                       // tile_dim0 = 6144
    g1[4] = 1;                                        // tile_dim1 = 1
    g1[5] = 6144;                                     // tensor_dim0_stride
    g1[6] = 0; g1[7] = 0;
    v4i_t z4 = {0, 0, 0, 0};
    v8i_t z8 = {0, 0, 0, 0, 0, 0, 0, 0};
    __builtin_amdgcn_tensor_load_to_lds(g0, g1, z4, z4, z8, 0);
  }
  __builtin_amdgcn_s_wait_tensorcnt(0);
  __syncthreads();
#else
  for (int i = threadIdx.x; i < 24576 / 8; i += blockDim.x)
    ((v8h*)wlds)[i] = ((const v8h*)wihf)[i];
  __syncthreads();
#endif
  const int wave = threadIdx.x >> 5, lane = threadIdx.x & 31;
  const int hi2 = lane >> 4, m = lane & 15;
  const int tile = blockIdx.x * 8 + wave;            // 8192 tiles
  const half_t* hrow = h16 + (size_t)(tile * 16 + m) * kH;
  v16h a[2];
#pragma unroll
  for (int ks = 0; ks < 2; ++ks)
    a[ks] = cat16(*(const v8h*)(hrow + ks * 32 + 8 * hi2),
                  *(const v8h*)(hrow + ks * 32 + 16 + 8 * hi2));
  const v16h* wih = (const v16h*)wlds;
  const v16h* whh = (const v16h*)(wlds + 12288);
#pragma unroll 1
  for (int j = 0; j < 4; ++j) {
    const int n = j * 16 + m;
    v8f cr = splat8(bih[n]),      cz = splat8(bih[64 + n]),  cn = splat8(bih[128 + n]);
    v8f dr = splat8(bhh[n]),      dz = splat8(bhh[64 + n]),  dn = splat8(bhh[128 + n]);
#pragma unroll
    for (int ks = 0; ks < 2; ++ks) {
      cr = wmma_f16(a[ks], wih[(ks * 12 + j    ) * 32 + lane], cr);
      cz = wmma_f16(a[ks], wih[(ks * 12 + j + 4) * 32 + lane], cz);
      cn = wmma_f16(a[ks], wih[(ks * 12 + j + 8) * 32 + lane], cn);
      dr = wmma_f16(a[ks], whh[(ks * 12 + j    ) * 32 + lane], dr);
      dz = wmma_f16(a[ks], whh[(ks * 12 + j + 4) * 32 + lane], dz);
      dn = wmma_f16(a[ks], whh[(ks * 12 + j + 8) * 32 + lane], dn);
    }
#pragma unroll
    for (int i = 0; i < 8; ++i) {
      const size_t idx = (size_t)(tile * 16 + i + 8 * hi2) * kH + j * 16 + m;
      const float hold = h32[idx];
      const float r = sigmoidf(cr[i] + dr[i]);
      const float z = sigmoidf(cz[i] + dz[i]);
      const float nn = tanhf(cn[i] + r * dn[i]);
      const float hv = (1.f - z) * nn + z * hold;
      h32[idx] = hv; h16[idx] = (half_t)hv;
    }
  }
}

// ================= Set2Set LSTM cell (B=4096 rows) =================
__global__ __launch_bounds__(256) void lstm_kernel(
    const half_t* __restrict__ qs16, half_t* __restrict__ hl16, float* __restrict__ hl32,
    float* __restrict__ cl, const half_t* __restrict__ wihf, const half_t* __restrict__ whhf,
    const float* __restrict__ bih, const float* __restrict__ bhh) {
  const int wave = threadIdx.x >> 5, lane = threadIdx.x & 31;
  const int hi2 = lane >> 4, m = lane & 15;
  const int tile = blockIdx.x * 8 + wave;            // 256 tiles
  const half_t* qrow = qs16 + (size_t)(tile * 16 + m) * 128;
  const half_t* hrow = hl16 + (size_t)(tile * 16 + m) * kH;
  v16h aq[4], ah[2];
#pragma unroll
  for (int ks = 0; ks < 4; ++ks)
    aq[ks] = cat16(*(const v8h*)(qrow + ks * 32 + 8 * hi2),
                   *(const v8h*)(qrow + ks * 32 + 16 + 8 * hi2));
#pragma unroll
  for (int ks = 0; ks < 2; ++ks)
    ah[ks] = cat16(*(const v8h*)(hrow + ks * 32 + 8 * hi2),
                   *(const v8h*)(hrow + ks * 32 + 16 + 8 * hi2));
  const v16h* wih = (const v16h*)wihf;
  const v16h* whh = (const v16h*)whhf;
#pragma unroll 1
  for (int j = 0; j < 4; ++j) {
    const int n = j * 16 + m;
    v8f ci = splat8(bih[n]       + bhh[n]);
    v8f cf = splat8(bih[64 + n]  + bhh[64 + n]);
    v8f cg = splat8(bih[128 + n] + bhh[128 + n]);
    v8f co = splat8(bih[192 + n] + bhh[192 + n]);
#pragma unroll
    for (int ks = 0; ks < 4; ++ks) {
      ci = wmma_f16(aq[ks], wih[(ks * 16 + j     ) * 32 + lane], ci);
      cf = wmma_f16(aq[ks], wih[(ks * 16 + j + 4 ) * 32 + lane], cf);
      cg = wmma_f16(aq[ks], wih[(ks * 16 + j + 8 ) * 32 + lane], cg);
      co = wmma_f16(aq[ks], wih[(ks * 16 + j + 12) * 32 + lane], co);
    }
#pragma unroll
    for (int ks = 0; ks < 2; ++ks) {
      ci = wmma_f16(ah[ks], whh[(ks * 16 + j     ) * 32 + lane], ci);
      cf = wmma_f16(ah[ks], whh[(ks * 16 + j + 4 ) * 32 + lane], cf);
      cg = wmma_f16(ah[ks], whh[(ks * 16 + j + 8 ) * 32 + lane], cg);
      co = wmma_f16(ah[ks], whh[(ks * 16 + j + 12) * 32 + lane], co);
    }
#pragma unroll
    for (int i = 0; i < 8; ++i) {
      const size_t idx = (size_t)(tile * 16 + i + 8 * hi2) * kH + j * 16 + m;
      const float cold = cl[idx];
      const float ig = sigmoidf(ci[i]), fg = sigmoidf(cf[i]);
      const float gg = tanhf(cg[i]),    og = sigmoidf(co[i]);
      const float cnew = fg * cold + ig * gg;
      const float hnew = og * tanhf(cnew);
      cl[idx] = cnew; hl32[idx] = hnew; hl16[idx] = (half_t)hnew;
    }
  }
}

// ================= attention (segment softmax + weighted segment sum) =================
__global__ __launch_bounds__(256) void att_dot_kernel(
    const half_t* __restrict__ h16, const half_t* __restrict__ hl16,
    const int* __restrict__ batch, float* __restrict__ ebuf, float* __restrict__ mbuf) {
  const int wave = threadIdx.x >> 5, lane = threadIdx.x & 31;
  const int node = blockIdx.x * 8 + wave;
  const int b = batch[node];
  const half_t* hr = h16 + (size_t)node * kH;
  const half_t* qr = hl16 + (size_t)b * kH;
  float acc = (float)hr[lane] * (float)qr[lane] +
              (float)hr[32 + lane] * (float)qr[32 + lane];
#pragma unroll
  for (int o = 16; o >= 1; o >>= 1) acc += __shfl_xor(acc, o, 32);
  if (lane == 0) { ebuf[node] = acc; atomicMaxF(&mbuf[b], acc); }
}

__global__ void att_exp_kernel(const float* __restrict__ ebuf, const float* __restrict__ mbuf,
                               const int* __restrict__ batch, float* __restrict__ exbuf,
                               float* __restrict__ den) {
  const int i = blockIdx.x * blockDim.x + threadIdx.x;
  if (i >= kN) return;
  const int b = batch[i];
  const float ex = __expf(ebuf[i] - mbuf[b]);
  exbuf[i] = ex;
  atomicAdd(&den[b], ex);
}

__global__ __launch_bounds__(256) void att_scatter_kernel(
    const half_t* __restrict__ h16, const float* __restrict__ exbuf,
    const float* __restrict__ den, const int* __restrict__ batch, float* __restrict__ rg) {
  const int wave = threadIdx.x >> 5, lane = threadIdx.x & 31;
  const int node = blockIdx.x * 8 + wave;
  const int b = batch[node];
  const float a = exbuf[node] / den[b];
  atomicAdd(&rg[(size_t)b * kH + lane],      a * (float)h16[(size_t)node * kH + lane]);
  atomicAdd(&rg[(size_t)b * kH + 32 + lane], a * (float)h16[(size_t)node * kH + 32 + lane]);
}

__global__ void qstar_build_kernel(const float* __restrict__ hl32, const float* __restrict__ rg,
                                   half_t* __restrict__ qs16) {
  const int i = blockIdx.x * blockDim.x + threadIdx.x;
  if (i >= kB * 128) return;
  const int b = i >> 7, c = i & 127;
  const float v = (c < 64) ? hl32[(size_t)b * kH + c] : rg[(size_t)b * kH + c - 64];
  qs16[i] = (half_t)v;
}

// ================= final MLP: relu(q*@fc1+b) @ fc2 + b =================
__global__ __launch_bounds__(256) void final_kernel(
    const half_t* __restrict__ qs16, const half_t* __restrict__ fc1f,
    const float* __restrict__ fc1_b, const float* __restrict__ fc2_w,
    const float* __restrict__ fc2_b, float* __restrict__ out) {
  const int wave = threadIdx.x >> 5, lane = threadIdx.x & 31;
  const int hi2 = lane >> 4, m = lane & 15;
  const int tile = blockIdx.x * 8 + wave;            // 256 tiles
  const half_t* qrow = qs16 + (size_t)(tile * 16 + m) * 128;
  v16h a[4];
#pragma unroll
  for (int ks = 0; ks < 4; ++ks)
    a[ks] = cat16(*(const v8h*)(qrow + ks * 32 + 8 * hi2),
                  *(const v8h*)(qrow + ks * 32 + 16 + 8 * hi2));
  const v16h* bf = (const v16h*)fc1f;
  v8f c[4];
#pragma unroll
  for (int nt = 0; nt < 4; ++nt) c[nt] = splat8(fc1_b[nt * 16 + m]);
#pragma unroll
  for (int ks = 0; ks < 4; ++ks)
#pragma unroll
    for (int nt = 0; nt < 4; ++nt)
      c[nt] = wmma_f16(a[ks], bf[(ks * 4 + nt) * 32 + lane], c[nt]);
  float w2[4];
#pragma unroll
  for (int nt = 0; nt < 4; ++nt) w2[nt] = fc2_w[nt * 16 + m];
  const float bout = fc2_b[0];
#pragma unroll
  for (int i = 0; i < 8; ++i) {
    float p = 0.f;
#pragma unroll
    for (int nt = 0; nt < 4; ++nt) p += fmaxf(c[nt][i], 0.f) * w2[nt];
#pragma unroll
    for (int o = 8; o >= 1; o >>= 1) p += __shfl_xor(p, o, 16);
    if (m == 0) out[tile * 16 + i + 8 * hi2] = p + bout;
  }
}

// ================= host launcher =================
extern "C" void kernel_launch(void* const* d_in, const int* in_sizes, int n_in,
                              void* d_out, int out_size, void* d_ws, size_t ws_size,
                              hipStream_t stream) {
  (void)in_sizes; (void)n_in; (void)out_size; (void)ws_size;
  const float* x         = (const float*)d_in[0];
  const float* edge_attr = (const float*)d_in[1];
  const int*   edge_index= (const int*)d_in[2];
  const int*   batch     = (const int*)d_in[3];
  const float* w_e1      = (const float*)d_in[4];
  const float* b_e1      = (const float*)d_in[5];
  const float* w_e2      = (const float*)d_in[6];
  const float* b_e2      = (const float*)d_in[7];
  const float* root      = (const float*)d_in[8];
  const float* conv_bias = (const float*)d_in[9];
  const float* gru_wih   = (const float*)d_in[10];
  const float* gru_whh   = (const float*)d_in[11];
  const float* gru_bih   = (const float*)d_in[12];
  const float* gru_bhh   = (const float*)d_in[13];
  const float* lstm_wih  = (const float*)d_in[14];
  const float* lstm_whh  = (const float*)d_in[15];
  const float* lstm_bih  = (const float*)d_in[16];
  const float* lstm_bhh  = (const float*)d_in[17];
  const float* fc1_w     = (const float*)d_in[18];
  const float* fc1_b     = (const float*)d_in[19];
  const float* fc2_w     = (const float*)d_in[20];
  const float* fc2_b     = (const float*)d_in[21];

  char* ws = (char*)d_ws;
  half_t* w2r   = (half_t*)(ws + OFF_W2R);
  half_t* rootf = (half_t*)(ws + OFF_ROOT);
  half_t* gwih  = (half_t*)(ws + OFF_GWIH);
  half_t* gwhh  = (half_t*)(ws + OFF_GWHH);
  half_t* lwih  = (half_t*)(ws + OFF_LWIH);
  half_t* lwhh  = (half_t*)(ws + OFF_LWHH);
  half_t* fc1f  = (half_t*)(ws + OFF_FC1);
  float*  msum  = (float*)(ws + OFF_MSUM);
  float*  cnt   = (float*)(ws + OFF_CNT);
  float*  h32   = (float*)(ws + OFF_H32);
  half_t* h16   = (half_t*)(ws + OFF_H16);
  float*  hl32  = (float*)(ws + OFF_HL32);
  float*  cl    = (float*)(ws + OFF_CL);
  half_t* hl16  = (half_t*)(ws + OFF_HL16);
  half_t* qs16  = (half_t*)(ws + OFF_QS16);
  float*  ebuf  = (float*)(ws + OFF_EB);
  float*  exbuf = (float*)(ws + OFF_EX);
  float*  mbuf  = (float*)(ws + OFF_M);
  float*  den   = (float*)(ws + OFF_DEN);
  float*  rg    = (float*)(ws + OFF_RG);

  // --- pack weights into WMMA B-fragment layouts (f16) ---
  pack_w2r_kernel<<<520, 256, 0, stream>>>(w_e2, b_e2, w2r);
  pack_b_frags<<<8,   256, 0, stream>>>(root,     rootf, 1,  4,  16,  64,  64, 0);
  pack_b_frags<<<48,  256, 0, stream>>>(gru_wih,  gwih,  2, 12,  64, 192,  64, 1);
  pack_b_frags<<<48,  256, 0, stream>>>(gru_whh,  gwhh,  2, 12,  64, 192,  64, 1);
  pack_b_frags<<<128, 256, 0, stream>>>(lstm_wih, lwih,  4, 16, 128, 256, 128, 1);
  pack_b_frags<<<64,  256, 0, stream>>>(lstm_whh, lwhh,  2, 16,  64, 256,  64, 1);
  pack_b_frags<<<32,  256, 0, stream>>>(fc1_w,    fc1f,  4,  4, 128,  64,  64, 0);

  // --- init runtime state ---
  (void)hipMemsetAsync(ws + OFF_MSUM, 0, (size_t)kN * 64 * 4 + (size_t)kN * 4, stream); // msum + cnt
  (void)hipMemsetAsync(ws + OFF_HL32, 0, OFF_EB - OFF_HL32, stream);                    // hl32, cl, hl16, qs16

  // --- NNConv (WMMA) + scatter-mean ---
  edge_msg_kernel<<<1024, 128, 0, stream>>>(x, edge_attr, edge_index, w_e1, b_e1, w2r, msum, cnt);
  node_update_kernel<<<1024, 256, 0, stream>>>(x, msum, cnt, rootf, conv_bias, h32, h16);

  // --- 3 GRU steps (in-place) ---
  for (int s = 0; s < 3; ++s)
    gru_step_kernel<<<1024, 256, 0, stream>>>(h32, h16, gwih, gru_bih, gru_bhh);

  // --- Set2Set, 3 processing steps ---
  for (int t = 0; t < 3; ++t) {
    lstm_kernel<<<32, 256, 0, stream>>>(qs16, hl16, hl32, cl, lwih, lwhh, lstm_bih, lstm_bhh);
    fill_f32_kernel<<<16, 256, 0, stream>>>(mbuf, -1e30f, kB);
    (void)hipMemsetAsync(ws + OFF_DEN, 0, (size_t)kB * 4 + (size_t)kB * 64 * 4, stream); // den + rg
    att_dot_kernel<<<kN / 8, 256, 0, stream>>>(h16, hl16, batch, ebuf, mbuf);
    att_exp_kernel<<<kN / 256, 256, 0, stream>>>(ebuf, mbuf, batch, exbuf, den);
    att_scatter_kernel<<<kN / 8, 256, 0, stream>>>(h16, exbuf, den, batch, rg);
    qstar_build_kernel<<<kB * 128 / 256, 256, 0, stream>>>(hl32, rg, qs16);
  }

  // --- readout MLP ---
  final_kernel<<<32, 256, 0, stream>>>(qs16, fc1f, fc1_b, fc2_w, fc2_b, (float*)d_out);
}
